// Net_40209483825636
// MI455X (gfx1250) — matrix operand order repaired and checked by reading, not compiled, not run on hardware
//
#include <hip/hip_runtime.h>
#include <math.h>

// ---------------------------------------------------------------------------
// Clifford FNO forward pass for gfx1250 (MI455X).
//   - 1x1 Clifford convs   -> f16 WMMA GEMM (v_wmma_f32_16x16x32_f16),
//     single-pass over activations (all col-tiles per block), A fragments
//     loaded directly from global as two contiguous float8 runs per lane,
//     B panel staged K-major in LDS; B fragment = column lane%16 of the panel.
//   - spectral blocks      -> partial separable DFT (only central 16^3 modes)
//   - gnorm                -> reduce + 1-thread 8x8 Cholesky + fused apply
//   - 5^3 Clifford convs   -> direct (memory-bound, <=32 channels)
// Activation layout: [voxel][channel], channel = blade*C + c (blade-major).
// Input pointer order assumes setup_inputs() dict insertion order.
// ---------------------------------------------------------------------------

#define NVOX 512000   // 80^3
#define MDIM 80
#define NMODE 16      // 2*MODES per axis
#define HCH 25        // hidden channels per blade
#define CT 200        // 8*HCH

typedef __attribute__((ext_vector_type(16))) _Float16 v16h;
typedef __attribute__((ext_vector_type(8)))  _Float16 h8;
typedef __attribute__((ext_vector_type(8)))  float    v8f;

// S[i][j][m] has exactly one nonzero (+-1): store mode index and sign.
__device__ __constant__ int   c_m[64] = {
  0,1,2,3,4,5,6,7,
  1,0,4,5,2,3,7,6,
  2,4,0,6,1,7,3,5,
  3,5,6,0,7,1,2,4,
  4,2,1,7,0,6,5,3,
  5,3,7,1,6,0,4,2,
  6,7,3,2,5,4,0,1,
  7,6,5,4,3,2,1,0 };
__device__ __constant__ float c_s[64] = {
  1, 1, 1, 1,-1,-1,-1,-1,
  1, 1,-1,-1, 1, 1,-1,-1,
  1, 1, 1,-1,-1, 1, 1, 1,
  1, 1, 1, 1,-1,-1,-1,-1,
  1, 1,-1, 1, 1,-1, 1, 1,
  1, 1,-1,-1, 1, 1,-1,-1,
  1, 1, 1,-1,-1, 1, 1, 1,
  1, 1,-1, 1, 1,-1, 1, 1 };

__device__ __forceinline__ float gelu_exact(float x) {
  return 0.5f * x * (1.0f + erff(0.70710678118654752f * x));
}

// ---------------------------------------------------------------------------
// WMMA GEMM:  Cout[row, o] = act( sum_k A[row,k] * W[o,k] + bias[o] )
// A: f32 [Nrows x K] row-major (K % 8 == 0, rows 32B-aligned).
// Wpt: f16 tiled pack [Kpad][NT*16] (k-major), zero padded.
// Block: 128 thr = 4 waves, 64 rows x NT*16 cols; activations read once.
// ---------------------------------------------------------------------------
template<int NT>
__global__ __launch_bounds__(128)
void gemm_cliff(const float* __restrict__ A, const _Float16* __restrict__ Wpt,
                const float* __restrict__ bias, float* __restrict__ Cout,
                int K, int Kpad, int Nout, int actGelu)
{
  __shared__ _Float16 Bs[32][NT * 16];   // K-major panel for one 32-chunk
  const int tid  = threadIdx.x;
  const int wv   = tid >> 5;
  const int lane = tid & 31;
  const int rowBase = blockIdx.x * 64;
  const int myrow   = rowBase + wv * 16 + (lane & 15);
  const int g       = (lane < 16) ? 0 : 8;   // half-wave K offset (A layout)

  v8f acc[NT];
  #pragma unroll
  for (int t = 0; t < NT; ++t) acc[t] = v8f{0.f,0.f,0.f,0.f,0.f,0.f,0.f,0.f};

  const int nchunk = Kpad >> 5;
  const float* arow = A + (size_t)myrow * K;

  for (int ch = 0; ch < nchunk; ++ch) {
    const int k0 = ch << 5;

    // ---- stage B panel chunk (straight vector memcpy, pack layout == LDS) --
    {
      const uint4* src = (const uint4*)(Wpt + (size_t)k0 * (NT * 16));
      uint4* dst = (uint4*)&Bs[0][0];
      #pragma unroll 4
      for (int t = tid; t < (32 * NT * 16) / 8; t += 128) dst[t] = src[t];
    }
    __syncthreads();

    // ---- A fragment: two contiguous 8xf32 runs per lane -------------------
    v8f a0 = {}, a1 = {};
    if (k0 + g < K)      a0 = *(const v8f*)(arow + k0 + g);
    if (k0 + 16 + g < K) a1 = *(const v8f*)(arow + k0 + 16 + g);
    v16h af;
    #pragma unroll
    for (int e = 0; e < 8; ++e) {
      af[e]     = (_Float16)a0[e];
      af[8 + e] = (_Float16)a1[e];
    }
    if (ch + 1 < nchunk)   // gfx1250 global_prefetch_b8
      __builtin_prefetch(arow + k0 + 32 + g, 0, 1);

    // ---- NT wmma, B fragment = column lane%16 of LDS panel, k = g+e | 16+g+e --
    #pragma unroll
    for (int t = 0; t < NT; ++t) {
      const int bn = t * 16 + (lane & 15);
      v16h bf;
      #pragma unroll
      for (int e = 0; e < 8; ++e) { bf[e] = Bs[g + e][bn]; bf[8 + e] = Bs[16 + g + e][bn]; }
      acc[t] = __builtin_amdgcn_wmma_f32_16x16x32_f16(false, af, false, bf,
                                                      (short)0, acc[t],
                                                      false, false);
    }
    __syncthreads();
  }

  // ---- epilogue: D layout VGPR r -> M = r | r+8, N = lane%16 --------------
  const int col  = lane & 15;
  const int moff = (lane < 16) ? 0 : 8;
  #pragma unroll
  for (int t = 0; t < NT; ++t) {
    int gcol = t * 16 + col;
    if (gcol < Nout) {
      const float bv = bias[gcol];
      #pragma unroll
      for (int r = 0; r < 8; ++r) {
        int grow = rowBase + wv * 16 + r + moff;
        float v = acc[t][r] + bv;
        if (actGelu) v = gelu_exact(v);
        Cout[(size_t)grow * Nout + gcol] = v;
      }
    }
  }
}

// Pack Clifford-expanded 1x1 weights, K-major tiled:
//   Wpt[kglob][col] = S-sign * w[m(i,j),o,c],  col=i*O+o, kglob=j*C+c
__global__ void pack_w1x1(const float* __restrict__ w, _Float16* __restrict__ Wp,
                          int O, int C, int NC /*NT*16*/, int Kpad)
{
  int idx = blockIdx.x * blockDim.x + threadIdx.x;
  if (idx >= Kpad * NC) return;
  int kglob = idx / NC, col = idx % NC;
  float v = 0.f;
  if (col < 8 * O && kglob < 8 * C) {
    int i = col / O,   o = col % O;
    int j = kglob / C, c = kglob % C;
    v = c_s[i * 8 + j] * w[(size_t)(c_m[i * 8 + j] * O + o) * C + c];
  }
  Wp[idx] = (_Float16)v;
}

__global__ void pack_bias(const float* __restrict__ b, float* __restrict__ bx,
                          int O, int rowpad)
{
  int idx = blockIdx.x * blockDim.x + threadIdx.x;
  if (idx >= rowpad) return;
  bx[idx] = (idx < 8 * O) ? b[idx] : 0.f;   // b is (8,O) -> flat i*O+o
}

// ---------------------------------------------------------------------------
// Direct 5x5x5 Clifford conv (pad=2). in:[vox][8C] out:[vox][8O]
// ---------------------------------------------------------------------------
__global__ void cconv5_kernel(const float* __restrict__ in,
                              const float* __restrict__ w,
                              const float* __restrict__ bias,
                              float* __restrict__ out, int C, int O)
{
  int idx = blockIdx.x * blockDim.x + threadIdx.x;
  const int Ot = 8 * O, Ctot = 8 * C;
  if (idx >= NVOX * Ot) return;
  int og  = idx % Ot;
  int vox = idx / Ot;
  int i = og / O, o = og % O;
  int z = vox % MDIM, y = (vox / MDIM) % MDIM, x = vox / (MDIM * MDIM);

  float acc = bias[i * O + o];
  for (int j = 0; j < 8; ++j) {
    float s = c_s[i * 8 + j];
    const float* wm = w + (size_t)(c_m[i * 8 + j] * O + o) * C * 125;
    for (int c = 0; c < C; ++c) {
      const float* wc = wm + (size_t)c * 125;
      int cin = j * C + c;
      for (int dx = 0; dx < 5; ++dx) {
        int nx = x + dx - 2; if (nx < 0 || nx >= MDIM) continue;
        for (int dy = 0; dy < 5; ++dy) {
          int ny = y + dy - 2; if (ny < 0 || ny >= MDIM) continue;
          size_t rb = ((size_t)(nx * MDIM + ny)) * MDIM;
          for (int dz = 0; dz < 5; ++dz) {
            int nz = z + dz - 2; if (nz < 0 || nz >= MDIM) continue;
            acc += s * wc[(dx * 5 + dy) * 5 + dz] *
                   in[(rb + nz) * Ctot + cin];
          }
        }
      }
    }
  }
  out[(size_t)vox * Ot + og] = acc;
}

// ---------------------------------------------------------------------------
// gnorm: stats (mean + second moments) -> Cholesky/inverse -> fused apply
// ---------------------------------------------------------------------------
__global__ void zero_stats(float* acc) {
  int t = threadIdx.x; if (t < 44) acc[t] = 0.f;
}

__global__ __launch_bounds__(256)
void gnorm_stats(const float* __restrict__ in, int C, float* __restrict__ acc)
{
  const int Ctot = 8 * C;
  float loc[44];
  for (int q = 0; q < 44; ++q) loc[q] = 0.f;

  for (int vox = blockIdx.x * blockDim.x + threadIdx.x; vox < NVOX;
       vox += gridDim.x * blockDim.x) {
    const float* p = in + (size_t)vox * Ctot;
    for (int c = 0; c < C; ++c) {
      float v[8];
      #pragma unroll
      for (int i = 0; i < 8; ++i) v[i] = p[i * C + c];
      #pragma unroll
      for (int i = 0; i < 8; ++i) loc[i] += v[i];
      #pragma unroll
      for (int i = 0; i < 8; ++i)
        #pragma unroll
        for (int j = 0; j <= i; ++j)
          loc[8 + (i * (i + 1)) / 2 + j] += v[i] * v[j];
    }
  }

  __shared__ float sb[256];
  for (int q = 0; q < 44; ++q) {
    sb[threadIdx.x] = loc[q];
    __syncthreads();
    for (int s = 128; s > 0; s >>= 1) {
      if (threadIdx.x < s) sb[threadIdx.x] += sb[threadIdx.x + s];
      __syncthreads();
    }
    if (threadIdx.x == 0) atomicAdd(&acc[q], sb[0]);
    __syncthreads();
  }
}

__global__ void gnorm_finalize(const float* __restrict__ acc, int C,
                               float* __restrict__ meanL)
{
  if (threadIdx.x != 0 || blockIdx.x != 0) return;
  const float n = (float)NVOX * (float)C;
  float mu[8];
  for (int i = 0; i < 8; ++i) mu[i] = acc[i] / n;
  float cov[8][8];
  int t = 0;
  for (int i = 0; i < 8; ++i)
    for (int j = 0; j <= i; ++j) {
      float cv = acc[8 + t] / n - mu[i] * mu[j];
      cov[i][j] = cov[j][i] = cv; ++t;
    }
  for (int i = 0; i < 8; ++i) cov[i][i] += 1e-5f;
  float L[8][8];
  for (int i = 0; i < 8; ++i) for (int j = 0; j < 8; ++j) L[i][j] = 0.f;
  for (int i = 0; i < 8; ++i)
    for (int j = 0; j <= i; ++j) {
      float s = cov[i][j];
      for (int k = 0; k < j; ++k) s -= L[i][k] * L[j][k];
      L[i][j] = (i == j) ? sqrtf(s) : s / L[j][j];
    }
  float Li[8][8];
  for (int i = 0; i < 8; ++i) for (int j = 0; j < 8; ++j) Li[i][j] = 0.f;
  for (int i = 0; i < 8; ++i) {
    Li[i][i] = 1.0f / L[i][i];
    for (int j = 0; j < i; ++j) {
      float s = 0.f;
      for (int k = j; k < i; ++k) s += L[i][k] * Li[k][j];
      Li[i][j] = -s / L[i][i];
    }
  }
  for (int i = 0; i < 8; ++i) meanL[i] = mu[i];
  for (int i = 0; i < 8; ++i)
    for (int j = 0; j < 8; ++j) meanL[8 + i * 8 + j] = Li[i][j];
}

__global__ void gnorm_apply(const float* __restrict__ in,
                            const float* __restrict__ meanL,
                            const float* __restrict__ nw,   // (8,8,C)
                            const float* __restrict__ nb,   // (8,C)
                            float* __restrict__ out, int C, int actGelu)
{
  int idx = blockIdx.x * blockDim.x + threadIdx.x;
  if (idx >= NVOX * C) return;
  int c = idx % C, vox = idx / C;
  const int Ctot = 8 * C;
  const float* p = in + (size_t)vox * Ctot;
  float v[8], y[8];
  #pragma unroll
  for (int i = 0; i < 8; ++i) v[i] = p[i * C + c] - meanL[i];
  #pragma unroll
  for (int i = 0; i < 8; ++i) {
    float s = 0.f;
    #pragma unroll
    for (int j = 0; j < 8; ++j)
      if (j <= i) s += meanL[8 + i * 8 + j] * v[j];
    y[i] = s;
  }
  float* q = out + (size_t)vox * Ctot;
  #pragma unroll
  for (int i = 0; i < 8; ++i) {
    float s = nb[i * C + c];
    #pragma unroll
    for (int j = 0; j < 8; ++j) s += nw[(i * 8 + j) * C + c] * y[j];
    if (actGelu) s = gelu_exact(s);
    q[i * C + c] = s;
  }
}

// ---------------------------------------------------------------------------
// Spectral block: partial DFT keeping only centered 16 modes per axis.
// fftshifted indices 32..47 <-> frequencies k = -8..7.
// ---------------------------------------------------------------------------
__global__ void make_twiddles(float2* fw, float2* iv)
{
  int t = blockIdx.x * blockDim.x + threadIdx.x;
  if (t >= NMODE * MDIM) return;
  int kk = t / MDIM, n = t % MDIM;
  float k = (float)(kk - 8);
  float ang = -6.283185307179586f * k * (float)n / (float)MDIM;
  fw[t] = make_float2(cosf(ang), sinf(ang));
  float a2 = -ang;
  iv[t] = make_float2(cosf(a2) / (float)MDIM, sinf(a2) / (float)MDIM);
}

__global__ void dft_fwd_z(const float* __restrict__ in, float2* __restrict__ F1,
                          const float2* __restrict__ tw)
{
  int idx = blockIdx.x * blockDim.x + threadIdx.x;
  if (idx >= CT * 6400 * NMODE) return;
  int kk = idx % NMODE;
  int xy = (idx / NMODE) % 6400;
  int ch = idx / (NMODE * 6400);
  float re = 0.f, im = 0.f;
  const float2* t = tw + kk * MDIM;
  const float* p = in + (size_t)xy * MDIM * CT + ch;
  for (int z = 0; z < MDIM; ++z) {
    float v = p[(size_t)z * CT];
    re += v * t[z].x; im += v * t[z].y;
  }
  F1[(size_t)ch * 102400 + xy * NMODE + kk] = make_float2(re, im);
}

__global__ void dft_fwd_y(const float2* __restrict__ F1, float2* __restrict__ F2,
                          const float2* __restrict__ tw)
{
  int idx = blockIdx.x * blockDim.x + threadIdx.x;
  if (idx >= CT * MDIM * NMODE * NMODE) return;
  int kz = idx % NMODE;
  int ky = (idx / NMODE) % NMODE;
  int x  = (idx / (NMODE * NMODE)) % MDIM;
  int ch = idx / (NMODE * NMODE * MDIM);
  float re = 0.f, im = 0.f;
  const float2* t = tw + ky * MDIM;
  const float2* p = F1 + (size_t)ch * 102400 + (size_t)x * MDIM * NMODE + kz;
  for (int y = 0; y < MDIM; ++y) {
    float2 u = p[(size_t)y * NMODE];
    re += u.x * t[y].x - u.y * t[y].y;
    im += u.x * t[y].y + u.y * t[y].x;
  }
  F2[(size_t)ch * 20480 + (x * NMODE + ky) * NMODE + kz] = make_float2(re, im);
}

__global__ void dft_fwd_x(const float2* __restrict__ F2, float2* __restrict__ F3,
                          const float2* __restrict__ tw)
{
  int idx = blockIdx.x * blockDim.x + threadIdx.x;
  if (idx >= CT * 4096) return;
  int p3 = idx % 4096;
  int ch = idx / 4096;
  int kx = p3 / 256, rem = p3 % 256;
  float re = 0.f, im = 0.f;
  const float2* t = tw + kx * MDIM;
  const float2* p = F2 + (size_t)ch * 20480 + rem;
  for (int x = 0; x < MDIM; ++x) {
    float2 u = p[(size_t)x * 256];
    re += u.x * t[x].x - u.y * t[x].y;
    im += u.x * t[x].y + u.y * t[x].x;
  }
  F3[(size_t)ch * 4096 + p3] = make_float2(re, im);
}

__global__ __launch_bounds__(256)
void spec_mix(const float2* __restrict__ F3, const float* __restrict__ w,
              float2* __restrict__ FM)
{
  int p = blockIdx.x;                    // 0..4095
  __shared__ float xr[CT], xi[CT];
  for (int c = threadIdx.x; c < CT; c += blockDim.x) {
    float2 v = F3[(size_t)c * 4096 + p];
    xr[c] = v.x; xi[c] = v.y;
  }
  __syncthreads();
  for (int og = threadIdx.x; og < CT; og += blockDim.x) {
    int i = og / HCH, o = og % HCH;
    float ar = 0.f, ai = 0.f;
    for (int j = 0; j < 8; ++j) {
      float s = c_s[i * 8 + j];
      const float* wp = w + ((size_t)(c_m[i * 8 + j] * HCH + o) * HCH) * 4096 + p;
      for (int c = 0; c < HCH; ++c) {
        float wv = s * wp[(size_t)c * 4096];
        ar += wv * xr[j * HCH + c];
        ai += wv * xi[j * HCH + c];
      }
    }
    FM[(size_t)og * 4096 + p] = make_float2(ar, ai);
  }
}

__global__ void dft_inv_x(const float2* __restrict__ FM, float2* __restrict__ G1,
                          const float2* __restrict__ tw)
{
  int idx = blockIdx.x * blockDim.x + threadIdx.x;
  if (idx >= CT * MDIM * 256) return;
  int rem = idx % 256;
  int x   = (idx / 256) % MDIM;
  int ch  = idx / (256 * MDIM);
  float re = 0.f, im = 0.f;
  const float2* p = FM + (size_t)ch * 4096 + rem;
  for (int kx = 0; kx < NMODE; ++kx) {
    float2 u = p[(size_t)kx * 256];
    float2 t = tw[kx * MDIM + x];
    re += u.x * t.x - u.y * t.y;
    im += u.x * t.y + u.y * t.x;
  }
  G1[(size_t)ch * 20480 + x * 256 + rem] = make_float2(re, im);
}

__global__ void dft_inv_y(const float2* __restrict__ G1, float2* __restrict__ G2,
                          const float2* __restrict__ tw)
{
  int idx = blockIdx.x * blockDim.x + threadIdx.x;
  if (idx >= CT * 6400 * NMODE) return;
  int kz = idx % NMODE;
  int y  = (idx / NMODE) % MDIM;
  int x  = (idx / (NMODE * MDIM)) % MDIM;
  int ch = idx / (NMODE * MDIM * MDIM);
  float re = 0.f, im = 0.f;
  const float2* p = G1 + (size_t)ch * 20480 + x * 256 + kz;
  for (int ky = 0; ky < NMODE; ++ky) {
    float2 u = p[(size_t)ky * NMODE];
    float2 t = tw[ky * MDIM + y];
    re += u.x * t.x - u.y * t.y;
    im += u.x * t.y + u.y * t.x;
  }
  G2[(size_t)ch * 102400 + (x * MDIM + y) * NMODE + kz] = make_float2(re, im);
}

__global__ void dft_inv_z_acc(const float2* __restrict__ G2,
                              const float2* __restrict__ tw,
                              float* __restrict__ out)
{
  int idx = blockIdx.x * blockDim.x + threadIdx.x;
  if (idx >= CT * NVOX) return;
  int ch  = idx % CT;
  int vox = idx / CT;
  int z   = vox % MDIM;
  int xy  = vox / MDIM;
  float re = 0.f;
  const float2* p = G2 + (size_t)ch * 102400 + (size_t)xy * NMODE;
  for (int kz = 0; kz < NMODE; ++kz) {
    float2 u = p[kz];
    float2 t = tw[kz * MDIM + z];
    re += u.x * t.x - u.y * t.y;
  }
  out[(size_t)vox * CT + ch] += re;
}

// ---------------------------------------------------------------------------
// small pointwise kernels
// ---------------------------------------------------------------------------
__global__ void perm_blades(const float* __restrict__ x, float* __restrict__ out)
{
  int vox = blockIdx.x * blockDim.x + threadIdx.x;
  if (vox >= NVOX) return;
  const int perm[8] = {7, 1, 2, 3, 4, 5, 6, 0};
  #pragma unroll
  for (int i = 0; i < 8; ++i)
    out[(size_t)vox * 8 + i] = x[(size_t)vox * 8 + perm[i]];
}

__global__ void final_mask(const float* __restrict__ h, const float* __restrict__ x,
                           float* __restrict__ out)
{
  int vox = blockIdx.x * blockDim.x + threadIdx.x;
  if (vox >= NVOX) return;
  out[vox] = h[(size_t)vox * 8 + 0] * x[(size_t)vox * 8 + 7];
}

// ---------------------------------------------------------------------------
// launcher
// ---------------------------------------------------------------------------
static inline dim3 g1(long long n, int b = 256) { return dim3((unsigned)((n + b - 1) / b)); }

extern "C" void kernel_launch(void* const* d_in, const int* in_sizes, int n_in,
                              void* d_out, int out_size, void* d_ws, size_t ws_size,
                              hipStream_t stream)
{
  (void)in_sizes; (void)n_in; (void)out_size; (void)ws_size;
  // --- inputs (setup_inputs dict insertion order) ---
  const float* X     = (const float*)d_in[0];
  const float* W01   = (const float*)d_in[1];
  const float* B01   = (const float*)d_in[2];
  const float* N01W  = (const float*)d_in[3];
  const float* N01B  = (const float*)d_in[4];
  const float* W02   = (const float*)d_in[5];
  const float* B02   = (const float*)d_in[6];
  const float* N02W  = (const float*)d_in[7];
  const float* N02B  = (const float*)d_in[8];
  const float* W03   = (const float*)d_in[9];
  const float* B03   = (const float*)d_in[10];
  const float* N03W  = (const float*)d_in[11];
  const float* N03B  = (const float*)d_in[12];
  const float* W1    = (const float*)d_in[13];
  const float* B1    = (const float*)d_in[14];
  const float* N1W   = (const float*)d_in[15];
  const float* N1B   = (const float*)d_in[16];
  const float* W2    = (const float*)d_in[17];
  const float* B2    = (const float*)d_in[18];
  const float* N2W   = (const float*)d_in[19];
  const float* N2B   = (const float*)d_in[20];
  const float* W3    = (const float*)d_in[21];
  const float* B3    = (const float*)d_in[22];
  const float* IN1W  = (const float*)d_in[23];
  const float* IN1B  = (const float*)d_in[24];
  const float* IN2W  = (const float*)d_in[25];
  const float* IN2B  = (const float*)d_in[26];
  const float* SPW0  = (const float*)d_in[27];
  const float* CW0   = (const float*)d_in[28];
  const float* CB0   = (const float*)d_in[29];
  const float* NW0   = (const float*)d_in[30];
  const float* NB0   = (const float*)d_in[31];
  const float* SPW1  = (const float*)d_in[32];
  const float* CW1   = (const float*)d_in[33];
  const float* CB1   = (const float*)d_in[34];
  const float* NW1   = (const float*)d_in[35];
  const float* NB1   = (const float*)d_in[36];
  const float* OUT1W = (const float*)d_in[37];
  const float* OUT1B = (const float*)d_in[38];
  const float* OUT2W = (const float*)d_in[39];
  const float* OUT2B = (const float*)d_in[40];
  float* OUT = (float*)d_out;

  // --- workspace carve (floats; f16 pack area kept 16B aligned) ---
  float* wsf = (float*)d_ws;
  size_t off = 0;
  auto fall = [&](size_t n) { float* p = wsf + off; off += (n + 3) & ~(size_t)3; return p; };
  float*  BIG0 = fall((size_t)NVOX * CT);
  float*  BIG1 = fall((size_t)NVOX * CT);
  float*  ENC0 = fall((size_t)NVOX * 32);
  float*  ENC1 = fall((size_t)NVOX * 32);
  float2* F1   = (float2*)fall((size_t)CT * 102400 * 2);
  float2* F2   = (float2*)fall((size_t)CT * 20480 * 2);   // reused as G1
  float2* F3   = (float2*)fall((size_t)CT * 4096 * 2);
  float2* FM   = (float2*)fall((size_t)CT * 4096 * 2);
  float2* TWF  = (float2*)fall((size_t)NMODE * MDIM * 2);
  float2* TWI  = (float2*)fall((size_t)NMODE * MDIM * 2);
  float*  STATS = fall(64);
  float*  MEANL = fall(80);
  float*  BXin1 = fall(208); float* BXin2 = fall(208);
  float*  BXc0  = fall(208); float* BXc1  = fall(208);
  float*  BXo1  = fall(208); float* BXo2  = fall(64);
  _Float16* HP = (_Float16*)(wsf + off);
  size_t hoff = 0;
  auto hall = [&](size_t n) { _Float16* p = HP + hoff; hoff += (n + 7) & ~(size_t)7; return p; };
  _Float16* WPin1 = hall((size_t)32  * 208);
  _Float16* WPin2 = hall((size_t)224 * 208);
  _Float16* WPc0  = hall((size_t)224 * 208);
  _Float16* WPc1  = hall((size_t)224 * 208);
  _Float16* WPo1  = hall((size_t)224 * 208);
  _Float16* WPo2  = hall((size_t)224 * 32);

  // --- prep: twiddles + weight packs (K-major tiled layout) ---
  make_twiddles<<<g1(NMODE * MDIM), 256, 0, stream>>>(TWF, TWI);
  pack_w1x1<<<g1(32 * 208),  256, 0, stream>>>(IN1W,  WPin1, HCH, 4,   208, 32);
  pack_w1x1<<<g1(224 * 208), 256, 0, stream>>>(IN2W,  WPin2, HCH, HCH, 208, 224);
  pack_w1x1<<<g1(224 * 208), 256, 0, stream>>>(CW0,   WPc0,  HCH, HCH, 208, 224);
  pack_w1x1<<<g1(224 * 208), 256, 0, stream>>>(CW1,   WPc1,  HCH, HCH, 208, 224);
  pack_w1x1<<<g1(224 * 208), 256, 0, stream>>>(OUT1W, WPo1,  HCH, HCH, 208, 224);
  pack_w1x1<<<g1(224 * 32),  256, 0, stream>>>(OUT2W, WPo2,  4,   HCH, 32,  224);
  pack_bias<<<g1(208), 256, 0, stream>>>(IN1B,  BXin1, HCH, 208);
  pack_bias<<<g1(208), 256, 0, stream>>>(IN2B,  BXin2, HCH, 208);
  pack_bias<<<g1(208), 256, 0, stream>>>(CB0,   BXc0,  HCH, 208);
  pack_bias<<<g1(208), 256, 0, stream>>>(CB1,   BXc1,  HCH, 208);
  pack_bias<<<g1(208), 256, 0, stream>>>(OUT1B, BXo1,  HCH, 208);
  pack_bias<<<g1(64),  256, 0, stream>>>(OUT2B, BXo2,  4,   64);

  auto gnorm = [&](const float* in, float* outp, const float* nw,
                   const float* nb, int C, int act) {
    zero_stats<<<1, 64, 0, stream>>>(STATS);
    gnorm_stats<<<512, 256, 0, stream>>>(in, C, STATS);
    gnorm_finalize<<<1, 1, 0, stream>>>(STATS, C, MEANL);
    gnorm_apply<<<g1((long long)NVOX * C), 256, 0, stream>>>(
        in, MEANL, nw, nb, outp, C, act);
  };

  // --- encoder ---
  perm_blades<<<g1(NVOX), 256, 0, stream>>>(X, ENC0);
  cconv5_kernel<<<g1((long long)NVOX * 8), 256, 0, stream>>>(ENC0, W01, B01, ENC1, 1, 1);
  gnorm(ENC1, ENC0, N01W, N01B, 1, 1);
  cconv5_kernel<<<g1((long long)NVOX * 16), 256, 0, stream>>>(ENC0, W02, B02, ENC1, 1, 2);
  gnorm(ENC1, ENC0, N02W, N02B, 2, 1);
  cconv5_kernel<<<g1((long long)NVOX * 32), 256, 0, stream>>>(ENC0, W03, B03, ENC1, 2, 4);
  gnorm(ENC1, ENC0, N03W, N03B, 4, 0);

  // --- FNO lift (WMMA GEMMs, all col-tiles per block) ---
  dim3 gg(NVOX / 64);
  gemm_cliff<13><<<gg, 128, 0, stream>>>(ENC0, WPin1, BXin1, BIG0, 32, 32, CT, 1);
  gemm_cliff<13><<<gg, 128, 0, stream>>>(BIG0, WPin2, BXin2, BIG1, CT, 224, CT, 1);

  // --- spectral blocks ---
  const float* spw[2] = {SPW0, SPW1};
  const _Float16* wpc[2] = {WPc0, WPc1};
  const float* bxc[2] = {BXc0, BXc1};
  const float* nww[2] = {NW0, NW1};
  const float* nbb[2] = {NB0, NB1};
  for (int b = 0; b < 2; ++b) {
    gemm_cliff<13><<<gg, 128, 0, stream>>>(BIG1, wpc[b], bxc[b], BIG0, CT, 224, CT, 0);
    dft_fwd_z<<<g1((long long)CT * 6400 * NMODE), 256, 0, stream>>>(BIG1, F1, TWF);
    dft_fwd_y<<<g1((long long)CT * MDIM * 256), 256, 0, stream>>>(F1, F2, TWF);
    dft_fwd_x<<<g1((long long)CT * 4096), 256, 0, stream>>>(F2, F3, TWF);
    spec_mix<<<4096, 256, 0, stream>>>(F3, spw[b], FM);
    dft_inv_x<<<g1((long long)CT * MDIM * 256), 256, 0, stream>>>(FM, F2, TWI);
    dft_inv_y<<<g1((long long)CT * 6400 * NMODE), 256, 0, stream>>>(F2, F1, TWI);
    dft_inv_z_acc<<<g1((long long)CT * NVOX), 256, 0, stream>>>(F1, TWI, BIG0);
    gnorm(BIG0, BIG1, nww[b], nbb[b], HCH, 1);
  }

  // --- FNO projection ---
  gemm_cliff<13><<<gg, 128, 0, stream>>>(BIG1, WPo1, BXo1, BIG0, CT, 224, CT, 1);
  gemm_cliff<2><<<gg, 128, 0, stream>>>(BIG0, WPo2, BXo2, ENC0, CT, 224, 32, 0);

  // --- decoder ---
  cconv5_kernel<<<g1((long long)NVOX * 16), 256, 0, stream>>>(ENC0, W1, B1, ENC1, 4, 2);
  gnorm(ENC1, ENC0, N1W, N1B, 2, 1);
  cconv5_kernel<<<g1((long long)NVOX * 8), 256, 0, stream>>>(ENC0, W2, B2, ENC1, 2, 1);
  gnorm(ENC1, ENC0, N2W, N2B, 1, 1);
  cconv5_kernel<<<g1((long long)NVOX * 8), 256, 0, stream>>>(ENC0, W3, B3, ENC1, 1, 1);

  final_mask<<<g1(NVOX), 256, 0, stream>>>(ENC1, X, OUT);
}
